// MultiHeadedAttention_33071248179244
// MI455X (gfx1250) — compile-verified, matrix operand order checked by reading, not run on hardware
//
#include <hip/hip_runtime.h>

// Problem constants (from reference)
#define BB   2
#define DMC  256
#define HH   4
#define MM   4096
#define KN   16
#define DIMC 64
#define MKSTR (MM * KN)   // 65536: channel stride in key/value
#define WPAD 264          // padded LDS row stride (bf16 elems): 528B = 4 dwords mod 64 banks

typedef __attribute__((ext_vector_type(16))) __bf16 v16bf;
typedef __attribute__((ext_vector_type(8)))  float  v8f;

union FragU {
  v16bf v;
  __bf16 h[16];
  uint4  q[2];
};

// Stage a 256x256 f32 weight matrix into LDS as bf16 (row stride WPAD).
// 256 threads, coalesced float4 reads, contiguous 8B LDS stores.
__device__ __forceinline__ void stage_weight(const float* __restrict__ W,
                                             __bf16 (*wb)[WPAD], int tid) {
  #pragma unroll 4
  for (int i = 0; i < 64; ++i) {
    const int idx = i * 256 + tid;   // float4 index; 64 float4 per row
    const int row = idx >> 6;
    const int c4  = (idx & 63) * 4;
    const float4 w4 = reinterpret_cast<const float4*>(W)[idx];
    wb[row][c4 + 0] = (__bf16)w4.x;
    wb[row][c4 + 1] = (__bf16)w4.y;
    wb[row][c4 + 2] = (__bf16)w4.z;
    wb[row][c4 + 3] = (__bf16)w4.w;
  }
}

// A-fragment (16x32 bf16 weight tile) from LDS: lane row M = base + l%16.
// K mapping (ISA 7.12.2, 16-bit A): lanes 0-15 hold K={0..7,16..23},
// lanes 16-31 hold K={8..15,24..31}. Two 16B ds_load_b128, no VALU.
__device__ __forceinline__ v16bf load_w_frag(const __bf16 (*wb)[WPAD],
                                             int row, int kk, int g) {
  FragU f;
  f.q[0] = *reinterpret_cast<const uint4*>(&wb[row][kk + g * 8]);
  f.q[1] = *reinterpret_cast<const uint4*>(&wb[row][kk + 16 + g * 8]);
  return f.v;
}

// B-fragment (32x16 bf16 activation tile): lane holds column N = l%16,
// mirrored per-lane K mapping. strideK = element stride between channels.
__device__ __forceinline__ v16bf load_in_frag(const float* __restrict__ base,
                                              long strideK, int kk, int g, int col) {
  FragU f;
  const float* p0 = base + (long)(kk + g * 8) * strideK + col;
  #pragma unroll
  for (int t = 0; t < 8; ++t) {
    f.h[t]     = (__bf16)p0[(long)t * strideK];
    f.h[8 + t] = (__bf16)p0[(long)(16 + t) * strideK];
  }
  return f.v;
}

// Project one 256-channel x 256-position tile into LDS (bf16, padded stride).
// 8 waves: wave w covers n-tiles {w, w+8}; per n-tile loops all 16 c-tiles.
__device__ __forceinline__ void proj_to_lds(const float* __restrict__ inb,
                                            const __bf16 (*wb)[WPAD],
                                            const float* __restrict__ bias,
                                            __bf16 (*kb)[WPAD],
                                            int wv, int lr, int g) {
  #pragma unroll
  for (int pass = 0; pass < 2; ++pass) {
    const int nt  = pass * 8 + wv;
    const int col = nt * 16 + lr;
    v16bf bfr[8];
    #pragma unroll
    for (int kk8 = 0; kk8 < 8; ++kk8)
      bfr[kk8] = load_in_frag(inb, MKSTR, kk8 * 32, g, col);
    for (int ct = 0; ct < 16; ++ct) {
      v8f acc = {};
      #pragma unroll
      for (int kk8 = 0; kk8 < 8; ++kk8) {
        v16bf a = load_w_frag(wb, ct * 16 + lr, kk8 * 32, g);
        acc = __builtin_amdgcn_wmma_f32_16x16x32_bf16(false, a, false, bfr[kk8],
                                                      (short)0, acc, false, false);
      }
      #pragma unroll
      for (int r = 0; r < 8; ++r) {
        const int row = ct * 16 + r + g * 8;           // D layout: M = r + 8*lanegroup
        kb[row][nt * 16 + lr] = (__bf16)(acc[r] + bias[row]);
      }
    }
  }
}

__global__ __launch_bounds__(256) void mha_stage1(
    const float* __restrict__ query, const float* __restrict__ key,
    const float* __restrict__ value,
    const float* __restrict__ Wq, const float* __restrict__ bq,
    const float* __restrict__ Wk, const float* __restrict__ bk,
    const float* __restrict__ Wv, const float* __restrict__ bv,
    float* __restrict__ xws, float* __restrict__ prob_out) {
  __shared__ __bf16 kbuf[DMC][WPAD];   // k/v projection tile (bf16), padded
  __shared__ __bf16 wbuf[DMC][WPAD];   // staged weight matrix (bf16), padded
  __shared__ float  qbuf[DMC][17];     // q projection tile, padded
  __shared__ float  pbuf[HH][16][KN];  // softmax probabilities

  const int tid  = threadIdx.x;
  const int lane = tid & 31;
  const int wv   = tid >> 5;       // wave 0..7
  const int lr   = lane & 15;
  const int g    = lane >> 4;

  const int blk = blockIdx.x;      // 0..511
  const int b   = blk >> 8;        // 256 m-blocks per batch
  const int m0  = (blk & 255) * 16;

  // Prefetch late-consumed operands toward L2/WGP$ (global_prefetch_b8):
  // the value tile row for this thread's channel, and this thread's Wv row.
  {
    const float* vrow = value + (long)b * DMC * MKSTR + (long)tid * MKSTR + m0 * KN;
    __builtin_prefetch(vrow, 0, 3);
    __builtin_prefetch(vrow + 128, 0, 3);
    __builtin_prefetch(vrow + 192, 0, 3);
    __builtin_prefetch(Wv + tid * DMC, 0, 3);
    __builtin_prefetch(Wv + tid * DMC + 128, 0, 3);
  }

  // ---- stage Wq, then q projection: 256x16, each wave owns 2 c-tiles ----
  stage_weight(Wq, wbuf, tid);
  __syncthreads();
  {
    const float* qbase = query + (long)b * DMC * MM + m0;  // (ci,mi)->qbase[ci*MM+mi]
    v16bf bfr[8];
    #pragma unroll
    for (int kk8 = 0; kk8 < 8; ++kk8)
      bfr[kk8] = load_in_frag(qbase, MM, kk8 * 32, g, lr);
    #pragma unroll
    for (int i = 0; i < 2; ++i) {
      const int ct = wv * 2 + i;
      v8f acc = {};
      #pragma unroll
      for (int kk8 = 0; kk8 < 8; ++kk8) {
        v16bf a = load_w_frag(wbuf, ct * 16 + lr, kk8 * 32, g);
        acc = __builtin_amdgcn_wmma_f32_16x16x32_bf16(false, a, false, bfr[kk8],
                                                      (short)0, acc, false, false);
      }
      #pragma unroll
      for (int r = 0; r < 8; ++r) {
        const int row = ct * 16 + r + g * 8;
        qbuf[row][lr] = acc[r] + bq[row];
      }
    }
  }
  __syncthreads();

  // ---- stage Wk, k projection -> kbuf ----
  stage_weight(Wk, wbuf, tid);
  __syncthreads();
  proj_to_lds(key + (long)b * DMC * MKSTR + m0 * KN, wbuf, bk, kbuf, wv, lr, g);
  __syncthreads();

  // ---- scores + softmax (threads 0-63) while remaining work stages Wv ----
  if (tid < 64) {
    const int h = tid & 3, mi = tid >> 2;
    float s[KN];
    float mx = -1e30f;
    #pragma unroll
    for (int kn = 0; kn < KN; ++kn) {
      float acc = 0.f;
      #pragma unroll
      for (int d = 0; d < DIMC; ++d) {
        const int c = d * HH + h;                      // channel c -> (d=c/H, h=c%H)
        acc += qbuf[c][mi] * (float)kbuf[c][mi * KN + kn];
      }
      s[kn] = acc * 0.125f;                            // 1/sqrt(64)
      mx = fmaxf(mx, s[kn]);
    }
    float sum = 0.f;
    #pragma unroll
    for (int kn = 0; kn < KN; ++kn) { s[kn] = __expf(s[kn] - mx); sum += s[kn]; }
    const float inv = 1.f / sum;
    #pragma unroll
    for (int kn = 0; kn < KN; ++kn) pbuf[h][mi][kn] = s[kn] * inv;
  }
  stage_weight(Wv, wbuf, tid);   // disjoint LDS region; overlaps softmax
  __syncthreads();               // kbuf reads done; pbuf + Wv visible

  // ---- prob_sum over heads: (b, m, kn) ----
  {
    const int mi = tid >> 4, kn = tid & 15;
    const float ps = pbuf[0][mi][kn] + pbuf[1][mi][kn] +
                     pbuf[2][mi][kn] + pbuf[3][mi][kn];
    prob_out[((long)b * MM + m0 + mi) * KN + kn] = ps;
  }

  // ---- v projection -> kbuf (reuse tile) ----
  proj_to_lds(value + (long)b * DMC * MKSTR + m0 * KN, wbuf, bv, kbuf, wv, lr, g);
  __syncthreads();

  // ---- x[c, m] = sum_kn prob[h][m][kn] * v[c][m,kn]  -> workspace ----
  #pragma unroll
  for (int it = 0; it < 16; ++it) {
    const int c  = (tid >> 4) + it * 16;
    const int mi = tid & 15;
    const int h  = c & 3;
    float acc = 0.f;
    #pragma unroll
    for (int kn = 0; kn < KN; ++kn)
      acc += pbuf[h][mi][kn] * (float)kbuf[c][mi * KN + kn];
    xws[((long)b * DMC + c) * MM + m0 + mi] = acc;
  }
}

__global__ __launch_bounds__(256) void mha_stage2(
    const float* __restrict__ xws, const float* __restrict__ Wm,
    const float* __restrict__ bm, float* __restrict__ out) {
  __shared__ __bf16 wbuf[DMC][WPAD];

  const int tid  = threadIdx.x;
  const int lane = tid & 31;
  const int wv   = tid >> 5;
  const int lr   = lane & 15;
  const int g    = lane >> 4;

  const int blk = blockIdx.x;        // 128 blocks: 64 positions each
  const int b   = blk >> 6;
  const int n0  = (blk & 63) * 64;
  const int nt  = wv & 3;            // 4 n-tiles, 2 waves each
  const int ct0 = (wv >> 2) * 8;     // 8 c-tiles per wave

  __builtin_prefetch(Wm + tid * DMC, 0, 3);
  __builtin_prefetch(Wm + tid * DMC + 128, 0, 3);

  stage_weight(Wm, wbuf, tid);

  const float* inb = xws + (long)b * DMC * MM + n0 + nt * 16;
  v16bf bfr[8];
  #pragma unroll
  for (int kk8 = 0; kk8 < 8; ++kk8)
    bfr[kk8] = load_in_frag(inb, MM, kk8 * 32, g, lr);

  __syncthreads();

  for (int i = 0; i < 8; ++i) {
    const int ct = ct0 + i;
    v8f acc = {};
    #pragma unroll
    for (int kk8 = 0; kk8 < 8; ++kk8) {
      v16bf a = load_w_frag(wbuf, ct * 16 + lr, kk8 * 32, g);
      acc = __builtin_amdgcn_wmma_f32_16x16x32_bf16(false, a, false, bfr[kk8],
                                                    (short)0, acc, false, false);
    }
    #pragma unroll
    for (int r = 0; r < 8; ++r) {
      const int row = ct * 16 + r + g * 8;
      out[((long)b * DMC + row) * MM + n0 + nt * 16 + lr] = acc[r] + bm[row];
    }
  }
}

extern "C" void kernel_launch(void* const* d_in, const int* in_sizes, int n_in,
                              void* d_out, int out_size, void* d_ws, size_t ws_size,
                              hipStream_t stream) {
  const float* query = (const float*)d_in[0];
  const float* key   = (const float*)d_in[1];
  const float* value = (const float*)d_in[2];
  const float* Wq    = (const float*)d_in[3];
  const float* bq    = (const float*)d_in[4];
  const float* Wk    = (const float*)d_in[5];
  const float* bk    = (const float*)d_in[6];
  const float* Wv    = (const float*)d_in[7];
  const float* bv    = (const float*)d_in[8];
  const float* Wm    = (const float*)d_in[9];
  const float* bm    = (const float*)d_in[10];

  float* out      = (float*)d_out;                       // (B, DM, M, 1)
  float* prob_out = out + (long)BB * DMC * MM;           // (B, M, K)
  float* xws      = (float*)d_ws;                        // (B, DM, M) f32, 8 MB

  hipLaunchKernelGGL(mha_stage1, dim3(BB * (MM / 16)), dim3(256), 0, stream,
                     query, key, value, Wq, bq, Wk, bk, Wv, bv, xws, prob_out);
  hipLaunchKernelGGL(mha_stage2, dim3(BB * (MM / 64)), dim3(256), 0, stream,
                     xws, Wm, bm, out);
}
